// ScalableSelfAttention_86131274154360
// MI455X (gfx1250) — matrix operand order repaired and checked by reading, not compiled, not run on hardware
//
#include <hip/hip_runtime.h>
#include <hip/hip_bf16.h>

typedef unsigned short ush;
typedef __attribute__((ext_vector_type(16))) __bf16 v16bf;
typedef __attribute__((ext_vector_type(8)))  float  v8f;

#define HEADS 8
#define DK 32
#define BATCH 8
#define HWD 56
#define NQ 3136       // 56*56
#define MKV 784       // 28*28
#define CH 256
#define KPAD 800      // MKV padded to multiple of 32
// (1/sqrt(DK)) * log2(e): fold softmax scale + base-2 conversion into Q
#define SCALE_LOG2E 0.2550656f

__device__ __forceinline__ ush f2bf(float f) {
  unsigned u = __float_as_uint(f);
  u += 0x7FFFu + ((u >> 16) & 1u);   // round-to-nearest-even
  return (ush)(u >> 16);
}

// Load 16 bf16 (two 16-byte chunks) into a WMMA operand vector.
__device__ __forceinline__ v16bf ld16(const ush* p0, const ush* p1) {
  union { uint4 u[2]; v16bf v; } t;
  t.u[0] = *reinterpret_cast<const uint4*>(p0);
  t.u[1] = *reinterpret_cast<const uint4*>(p1);
  return t.v;
}

__device__ __forceinline__ v16bf ones16() {
  union { uint4 u[2]; v16bf v; } t;
  uint4 q; q.x = q.y = q.z = q.w = 0x3F803F80u;   // bf16 1.0 pairs
  t.u[0] = q; t.u[1] = q;
  return t.v;
}

// ---------------- helper conversion kernels ----------------

__global__ void cvt_f32_bf16(const float* __restrict__ in, ush* __restrict__ out, int n) {
  int i = blockIdx.x * 256 + threadIdx.x;
  if (i < n) out[i] = f2bf(in[i]);
}

// W[k,n] (row-major KxN) -> Wt[n,k] bf16 (N-major, so B operand loads are contiguous)
__global__ void wtrans_bf16(const float* __restrict__ W, ush* __restrict__ Wt, int K, int N) {
  int i = blockIdx.x * 256 + threadIdx.x;        // i = n*K + k
  if (i < K * N) {
    int n = i / K, k = i - n * K;
    Wt[i] = f2bf(W[(size_t)k * N + n]);
  }
}

// Gather 2x2 stride-2 conv patches: pb[b*784+ij, tap*256+c] = x[b, 2i+th, 2j+tw, c]
__global__ void gather_patches(const float* __restrict__ x, ush* __restrict__ pb) {
  int i = blockIdx.x * 256 + threadIdx.x;        // over 6272*1024
  if (i < BATCH * MKV * 1024) {
    int p = i >> 10, q = i & 1023;
    int tap = q >> 8, c = q & 255;
    int b = p / MKV, ij = p - b * MKV;
    int oi = ij / 28, oj = ij - oi * 28;
    int si = oi * 2 + (tap >> 1), sj = oj * 2 + (tap & 1);
    pb[i] = f2bf(x[(((size_t)b * HWD + si) * HWD + sj) * CH + c]);
  }
}

// ---------------- WMMA GEMM: C[M,N] = alpha * A[M,K] @ Bt[N,K]^T (+bias) ----------------

template <bool BF16OUT, bool BIAS>
__global__ void gemm_wmma(const ush* __restrict__ A, const ush* __restrict__ Bt,
                          void* __restrict__ Cout, const float* __restrict__ bias,
                          int M, int N, int K, float alpha) {
  const int lane = threadIdx.x & 31;
  const int w    = threadIdx.x >> 5;
  const int row0 = (blockIdx.x * 8 + w) * 16;
  const int col0 = blockIdx.y * 32;
  if (row0 >= M) return;

  const int mrow   = row0 + (lane & 15);
  const int kchunk = lane >> 4;

  v8f c0 = {}, c1 = {};
  const ush* Ab = A  + (size_t)mrow * K + kchunk * 16;                   // 16 contiguous
  const ush* B0 = Bt + (size_t)(col0 + (lane & 15)) * K + kchunk * 16;   // 16 contiguous
  const ush* B1 = B0 + (size_t)16 * K;

#pragma unroll 4
  for (int k0 = 0; k0 < K; k0 += 32) {
    v16bf a  = ld16(Ab + k0, Ab + k0 + 8);
    v16bf b0 = ld16(B0 + k0, B0 + k0 + 8);
    v16bf b1 = ld16(B1 + k0, B1 + k0 + 8);
    c0 = __builtin_amdgcn_wmma_f32_16x16x32_bf16(false, a, false, b0, (short)0, c0, false, false);
    c1 = __builtin_amdgcn_wmma_f32_16x16x32_bf16(false, a, false, b1, (short)0, c1, false, false);
  }

  const int rbase = row0 + (lane >> 4) * 8;
  const int ccol  = col0 + (lane & 15);
  float bv0 = 0.0f, bv1 = 0.0f;
  if (BIAS) { bv0 = bias[ccol]; bv1 = bias[ccol + 16]; }

#pragma unroll
  for (int r = 0; r < 8; ++r) {
    float v0 = c0[r] * alpha + bv0;
    float v1 = c1[r] * alpha + bv1;
    size_t o0 = (size_t)(rbase + r) * N + ccol;
    if (BF16OUT) {
      ush* O = (ush*)Cout;
      O[o0] = f2bf(v0); O[o0 + 16] = f2bf(v1);
    } else {
      float* O = (float*)Cout;
      O[o0] = v0; O[o0 + 16] = v1;
    }
  }
}

// ---------------- fused attention, two-pass softmax, 32 Q rows per wave ----------------
// grid = (ceil(NQ/256), BATCH*HEADS). Block = 256 (8 waves), each wave owns 32 Q rows
// (two 16-row A tiles sharing every K/V B-operand load).
// LDS: K[800][32] bf16 + Vt[32][800] bf16 + per-wave P staging [8][2][16][64] bf16.
// Pass 1: QK WMMAs only, per-slot register max; one cross-lane reduction at the end.
// Pass 2: exp2(s - m) with fixed max; O and row-sum accumulated on the tensor core.

__global__ void attn_flash(const ush* __restrict__ Q, const ush* __restrict__ Kg,
                           const ush* __restrict__ Vg, ush* __restrict__ Og) {
  extern __shared__ ush smem[];
  ush* kls = smem;                      // row-major [m][d], m padded to 800
  ush* vls = smem + KPAD * 32;          // transposed [d][m]
  ush* pls = vls + 32 * KPAD;           // [wave][2][16][64]

  const int tid  = threadIdx.x;
  const int lane = tid & 31;
  const int w    = tid >> 5;
  const int bh   = blockIdx.y;
  const int b = bh >> 3, h = bh & 7;

  // --- K fill: async global->LDS copy (16B per lane), pad rows zeroed normally ---
  for (int c = tid; c < MKV * 4; c += 256) {        // 3136 16-byte chunks
    int m = c >> 2, d8 = (c & 3) * 8;
    unsigned loff = (unsigned)(uintptr_t)(kls + m * 32 + d8);
    unsigned long long ga =
        (unsigned long long)(uintptr_t)(Kg + ((size_t)(b * MKV + m)) * CH + h * DK + d8);
    asm volatile("global_load_async_to_lds_b128 %0, %1, off"
                 :: "v"(loff), "v"(ga) : "memory");
  }
  for (int i = tid; i < 16 * 32; i += 256) kls[MKV * 32 + i] = 0;

  // --- Vt fill: coalesced global reads (d fastest), scatter into LDS transpose ---
  for (int idx = tid; idx < MKV * 32; idx += 256) {
    int m = idx >> 5, d = idx & 31;
    vls[d * KPAD + m] = Vg[((size_t)(b * MKV + m)) * CH + h * DK + d];
  }
  for (int i = tid; i < 32 * 16; i += 256) {
    int d = i >> 4, m = MKV + (i & 15);
    vls[d * KPAD + m] = 0;
  }
  asm volatile("s_wait_asynccnt 0" ::: "memory");
  __syncthreads();

  // clamp tail-block waves instead of branching (EXEC all-ones for WMMA;
  // duplicate stores write identical values -> deterministic)
  int row0 = blockIdx.x * 256 + w * 32;
  if (row0 > NQ - 32) row0 = NQ - 32;

  const int colA   = lane & 15;
  const int kchunk = lane >> 4;

  const ush* qpA = Q + ((size_t)(b * NQ + row0 + colA)) * CH + h * DK + kchunk * 16;
  const ush* qpB = qpA + (size_t)16 * CH;
  v16bf qaA = ld16(qpA, qpA + 8);
  v16bf qaB = ld16(qpB, qpB + 8);
  const v16bf vone = ones16();

  // ================= pass 1: global row max (both row tiles) =================
  float mxA[8], mxB[8];
#pragma unroll
  for (int r = 0; r < 8; ++r) { mxA[r] = -1e30f; mxB[r] = -1e30f; }

#pragma unroll 1
  for (int t = 0; t < MKV / 64; ++t) {  // 12 rounds x 64 columns
    const int c0 = t * 64;
    v16bf kv[4];
#pragma unroll
    for (int tt = 0; tt < 4; ++tt) {
      const ush* kb = kls + (size_t)(c0 + tt * 16 + colA) * 32 + kchunk * 16;
      kv[tt] = ld16(kb, kb + 8);
    }
#pragma unroll
    for (int tt = 0; tt < 4; ++tt) {
      v8f z = {};
      v8f s = __builtin_amdgcn_wmma_f32_16x16x32_bf16(false, qaA, false, kv[tt], (short)0, z, false, false);
#pragma unroll
      for (int r = 0; r < 8; ++r) mxA[r] = fmaxf(mxA[r], s[r]);
    }
#pragma unroll
    for (int tt = 0; tt < 4; ++tt) {
      v8f z = {};
      v8f s = __builtin_amdgcn_wmma_f32_16x16x32_bf16(false, qaB, false, kv[tt], (short)0, z, false, false);
#pragma unroll
      for (int r = 0; r < 8; ++r) mxB[r] = fmaxf(mxB[r], s[r]);
    }
  }
  {   // tail columns 768..783
    const ush* kb = kls + (size_t)(768 + colA) * 32 + kchunk * 16;
    v16bf kv = ld16(kb, kb + 8);
    v8f z = {};
    v8f sA = __builtin_amdgcn_wmma_f32_16x16x32_bf16(false, qaA, false, kv, (short)0, z, false, false);
    v8f sB = __builtin_amdgcn_wmma_f32_16x16x32_bf16(false, qaB, false, kv, (short)0, z, false, false);
#pragma unroll
    for (int r = 0; r < 8; ++r) { mxA[r] = fmaxf(mxA[r], sA[r]); mxB[r] = fmaxf(mxB[r], sB[r]); }
  }
  // single cross-lane reduction over the 16 lanes sharing each row
#pragma unroll
  for (int r = 0; r < 8; ++r)
#pragma unroll
    for (int msk = 1; msk < 16; msk <<= 1) {
      mxA[r] = fmaxf(mxA[r], __shfl_xor(mxA[r], msk, 32));
      mxB[r] = fmaxf(mxB[r], __shfl_xor(mxB[r], msk, 32));
    }

  // ================= pass 2: softmax + PV, fixed max =================
  v8f oA0 = {}, oA1 = {}, lacA = {};
  v8f oB0 = {}, oB1 = {}, lacB = {};

  ush* pwA = pls + w * (2 * 16 * 64);
  ush* pwB = pwA + 16 * 64;
  const int rb8 = kchunk * 8;

#pragma unroll 1
  for (int t = 0; t < MKV / 64; ++t) {
    const int c0 = t * 64;
#pragma unroll
    for (int tt = 0; tt < 4; ++tt) {     // tile A scores -> staging
      const ush* kb = kls + (size_t)(c0 + tt * 16 + colA) * 32 + kchunk * 16;
      v16bf kv = ld16(kb, kb + 8);
      v8f z = {};
      v8f s = __builtin_amdgcn_wmma_f32_16x16x32_bf16(false, qaA, false, kv, (short)0, z, false, false);
#pragma unroll
      for (int r = 0; r < 8; ++r)
        pwA[(rb8 + r) * 64 + tt * 16 + colA] = f2bf(exp2f(s[r] - mxA[r]));
    }
#pragma unroll
    for (int tt = 0; tt < 4; ++tt) {     // tile B scores -> staging
      const ush* kb = kls + (size_t)(c0 + tt * 16 + colA) * 32 + kchunk * 16;
      v16bf kv = ld16(kb, kb + 8);
      v8f z = {};
      v8f s = __builtin_amdgcn_wmma_f32_16x16x32_bf16(false, qaB, false, kv, (short)0, z, false, false);
#pragma unroll
      for (int r = 0; r < 8; ++r)
        pwB[(rb8 + r) * 64 + tt * 16 + colA] = f2bf(exp2f(s[r] - mxB[r]));
    }

    __threadfence_block();   // order P stores (cross-lane) before A-layout reload
    const ush* paA = pwA + colA * 64 + kchunk * 16;
    const ush* paB = pwB + colA * 64 + kchunk * 16;
    v16bf pA0 = ld16(paA, paA + 8);
    v16bf pA1 = ld16(paA + 32, paA + 40);
    v16bf pB0 = ld16(paB, paB + 8);
    v16bf pB1 = ld16(paB + 32, paB + 40);

    const ush* vb = vls + (size_t)colA * KPAD + c0 + kchunk * 16;
    v16bf vv00 = ld16(vb, vb + 8);
    v16bf vv01 = ld16(vb + 16 * KPAD, vb + 16 * KPAD + 8);
    v16bf vv10 = ld16(vb + 32, vb + 40);
    v16bf vv11 = ld16(vb + 16 * KPAD + 32, vb + 16 * KPAD + 40);

    oA0  = __builtin_amdgcn_wmma_f32_16x16x32_bf16(false, pA0, false, vv00, (short)0, oA0,  false, false);
    oA1  = __builtin_amdgcn_wmma_f32_16x16x32_bf16(false, pA0, false, vv01, (short)0, oA1,  false, false);
    lacA = __builtin_amdgcn_wmma_f32_16x16x32_bf16(false, pA0, false, vone, (short)0, lacA, false, false);
    oA0  = __builtin_amdgcn_wmma_f32_16x16x32_bf16(false, pA1, false, vv10, (short)0, oA0,  false, false);
    oA1  = __builtin_amdgcn_wmma_f32_16x16x32_bf16(false, pA1, false, vv11, (short)0, oA1,  false, false);
    lacA = __builtin_amdgcn_wmma_f32_16x16x32_bf16(false, pA1, false, vone, (short)0, lacA, false, false);

    oB0  = __builtin_amdgcn_wmma_f32_16x16x32_bf16(false, pB0, false, vv00, (short)0, oB0,  false, false);
    oB1  = __builtin_amdgcn_wmma_f32_16x16x32_bf16(false, pB0, false, vv01, (short)0, oB1,  false, false);
    lacB = __builtin_amdgcn_wmma_f32_16x16x32_bf16(false, pB0, false, vone, (short)0, lacB, false, false);
    oB0  = __builtin_amdgcn_wmma_f32_16x16x32_bf16(false, pB1, false, vv10, (short)0, oB0,  false, false);
    oB1  = __builtin_amdgcn_wmma_f32_16x16x32_bf16(false, pB1, false, vv11, (short)0, oB1,  false, false);
    lacB = __builtin_amdgcn_wmma_f32_16x16x32_bf16(false, pB1, false, vone, (short)0, lacB, false, false);
    __threadfence_block();   // order P reload before next iteration's overwrite
  }

  // ---- tail: columns 768..783 (16 valid; pad P half with zeros) ----
  {
    const int c0 = MKV / 64 * 64;       // 768
    const ush* kb = kls + (size_t)(c0 + colA) * 32 + kchunk * 16;
    v16bf kv = ld16(kb, kb + 8);
    v8f z = {};
    v8f sA = __builtin_amdgcn_wmma_f32_16x16x32_bf16(false, qaA, false, kv, (short)0, z, false, false);
    v8f sB = __builtin_amdgcn_wmma_f32_16x16x32_bf16(false, qaB, false, kv, (short)0, z, false, false);
#pragma unroll
    for (int r = 0; r < 8; ++r) {
      pwA[(rb8 + r) * 64 + colA]      = f2bf(exp2f(sA[r] - mxA[r]));
      pwA[(rb8 + r) * 64 + 16 + colA] = 0;
      pwB[(rb8 + r) * 64 + colA]      = f2bf(exp2f(sB[r] - mxB[r]));
      pwB[(rb8 + r) * 64 + 16 + colA] = 0;
    }
    __threadfence_block();
    const ush* paA = pwA + colA * 64 + kchunk * 16;
    const ush* paB = pwB + colA * 64 + kchunk * 16;
    v16bf pA0 = ld16(paA, paA + 8);
    v16bf pB0 = ld16(paB, paB + 8);
    const ush* vb = vls + (size_t)colA * KPAD + c0 + kchunk * 16;   // cols 784..799 are zero
    v16bf vv00 = ld16(vb, vb + 8);
    v16bf vv01 = ld16(vb + 16 * KPAD, vb + 16 * KPAD + 8);
    oA0  = __builtin_amdgcn_wmma_f32_16x16x32_bf16(false, pA0, false, vv00, (short)0, oA0,  false, false);
    oA1  = __builtin_amdgcn_wmma_f32_16x16x32_bf16(false, pA0, false, vv01, (short)0, oA1,  false, false);
    lacA = __builtin_amdgcn_wmma_f32_16x16x32_bf16(false, pA0, false, vone, (short)0, lacA, false, false);
    oB0  = __builtin_amdgcn_wmma_f32_16x16x32_bf16(false, pB0, false, vv00, (short)0, oB0,  false, false);
    oB1  = __builtin_amdgcn_wmma_f32_16x16x32_bf16(false, pB0, false, vv01, (short)0, oB1,  false, false);
    lacB = __builtin_amdgcn_wmma_f32_16x16x32_bf16(false, pB0, false, vone, (short)0, lacB, false, false);
  }

  // epilogue: normalize (lac column = row sum) and store bf16 [b, n, h, d]
#pragma unroll
  for (int r = 0; r < 8; ++r) {
    float invA = 1.0f / lacA[r];
    float invB = 1.0f / lacB[r];
    int grow = row0 + kchunk * 8 + r;
    size_t baseA = ((size_t)(b * NQ + grow)) * CH + h * DK;
    size_t baseB = baseA + (size_t)16 * CH;
    Og[baseA + colA]      = f2bf(oA0[r] * invA);
    Og[baseA + 16 + colA] = f2bf(oA1[r] * invA);
    Og[baseB + colA]      = f2bf(oB0[r] * invB);
    Og[baseB + 16 + colA] = f2bf(oB1[r] * invB);
  }
}

// ---------------- host launcher ----------------

extern "C" void kernel_launch(void* const* d_in, const int* in_sizes, int n_in,
                              void* d_out, int out_size, void* d_ws, size_t ws_size,
                              hipStream_t stream) {
  (void)in_sizes; (void)n_in; (void)out_size; (void)ws_size;
  const float* x  = (const float*)d_in[0];
  const float* Wq = (const float*)d_in[1];
  const float* Wk = (const float*)d_in[2];
  const float* Wv = (const float*)d_in[3];
  const float* Wo = (const float*)d_in[4];
  const float* bo = (const float*)d_in[5];
  float* out = (float*)d_out;

  char* ws = (char*)d_ws;
  size_t off = 0;
  auto carve = [&](size_t bytes) -> void* {
    void* p = ws + off;
    off = (off + bytes + 255) & ~(size_t)255;
    return p;
  };
  const size_t NROW = (size_t)BATCH * NQ;        // 25088
  const size_t MROW = (size_t)BATCH * MKV;       // 6272
  ush* xb  = (ush*)carve(NROW * CH * 2);
  ush* pb  = (ush*)carve(MROW * 1024 * 2);       // conv patches (later reused as attn out)
  ush* qb  = (ush*)carve(NROW * CH * 2);
  ush* kbw = (ush*)carve(MROW * CH * 2);
  ush* vbw = (ush*)carve(MROW * CH * 2);
  ush* wqt = (ush*)carve(256ull * 256 * 2);
  ush* wkt = (ush*)carve(1024ull * 256 * 2);
  ush* wvt = (ush*)carve(1024ull * 256 * 2);
  ush* wot = (ush*)carve(256ull * 256 * 2);
  ush* ao  = pb;                                 // patches dead after K/V GEMMs

  const int nx = (int)(NROW * CH);
  cvt_f32_bf16<<<(nx + 255) / 256, 256, 0, stream>>>(x, xb, nx);
  wtrans_bf16<<<(256 * 256 + 255) / 256, 256, 0, stream>>>(Wq, wqt, 256, 256);
  wtrans_bf16<<<(1024 * 256 + 255) / 256, 256, 0, stream>>>(Wk, wkt, 1024, 256);
  wtrans_bf16<<<(1024 * 256 + 255) / 256, 256, 0, stream>>>(Wv, wvt, 1024, 256);
  wtrans_bf16<<<(256 * 256 + 255) / 256, 256, 0, stream>>>(Wo, wot, 256, 256);
  gather_patches<<<(int)((MROW * 1024 + 255) / 256), 256, 0, stream>>>(x, pb);

  gemm_wmma<true, false><<<dim3(196, 8), 256, 0, stream>>>(xb, wqt, qb, nullptr,
                                                           (int)NROW, CH, 256, SCALE_LOG2E);
  gemm_wmma<true, false><<<dim3(49, 8), 256, 0, stream>>>(pb, wkt, kbw, nullptr,
                                                          (int)MROW, CH, 1024, 1.0f);
  gemm_wmma<true, false><<<dim3(49, 8), 256, 0, stream>>>(pb, wvt, vbw, nullptr,
                                                          (int)MROW, CH, 1024, 1.0f);

  const size_t shmem = (size_t)(KPAD * 32 + 32 * KPAD + 8 * 2 * 16 * 64) * 2;  // 135168 B
  attn_flash<<<dim3(13, 64), 256, shmem, stream>>>(qb, kbw, vbw, ao);

  gemm_wmma<false, true><<<dim3(196, 8), 256, 0, stream>>>(ao, wot, out, bo,
                                                           (int)NROW, CH, 256, 1.0f);
}
